// GraphSage_58196806861274
// MI455X (gfx1250) — matrix operand order repaired and checked by reading, not compiled, not run on hardware
//
#include <hip/hip_runtime.h>
#include <hip/hip_bf16.h>

typedef __attribute__((ext_vector_type(16))) __bf16 v16bf;
typedef __attribute__((ext_vector_type(8)))  __bf16 v8bf;
typedef __attribute__((ext_vector_type(4)))  __bf16 v4bf;
typedef __attribute__((ext_vector_type(8)))  float  v8f;

#define D_IN        256
#define D_HALF      128
#define LDS_STRIDE  280   // bf16 elems; 560B/row: 16B-aligned, bank-rotates 12 banks/row

// ---------------------------------------------------------------------------
// f32 -> bf16 weight conversion (done once per launch; weights live in L2)
// ---------------------------------------------------------------------------
__global__ __launch_bounds__(256)
void cvt_bf16_kernel(const float* __restrict__ a, __bf16* __restrict__ o, int n) {
    const int i = blockIdx.x * 256 + threadIdx.x;
    if (i < n) o[i] = (__bf16)a[i];
}

// ---------------------------------------------------------------------------
// One wave per 16-row tile per output half.
//   1) gather + fanout-mean 16x256 panel into LDS (bf16)
//   2) 8 n-tiles x 8 k-tiles of v_wmma_f32_16x16x32_bf16 against W [128,256]
//   3) epilogue: relu(x+b) -> bf16 (agg layers), or x*rnorm[row]+b -> f32 (fc)
// ---------------------------------------------------------------------------
__global__ __launch_bounds__(32)
void sage_half_wmma_kernel(const void* __restrict__ srcv, int srcIsBf16,
                           const int* __restrict__ idx, int fanout,
                           const __bf16* __restrict__ W,      // [128,256] bf16 (row major)
                           const float* __restrict__ bias,    // [128]
                           const float* __restrict__ rnorm,   // null => relu+bf16 out
                           void* __restrict__ outv,
                           int outStride, int colBase)
{
    __shared__ __bf16 P[16 * LDS_STRIDE];

    const int lane    = threadIdx.x & 31;
    const int rowBase = blockIdx.x * 16;
    const float inv   = 1.0f / (float)fanout;

    // ---- stage 16x256 (mean over fanout) panel into LDS as bf16 ----
    const float*  srcF = (const float*)srcv;
    const __bf16* srcB = (const __bf16*)srcv;
    for (int e = lane; e < 16 * (D_IN / 4); e += 32) {
        const int r    = e >> 6;            // 64 4-wide chunks per row
        const int c4   = (e & 63) << 2;
        const int grow = rowBase + r;
        float ax = 0.f, ay = 0.f, az = 0.f, aw = 0.f;
        for (int t = 0; t < fanout; ++t) {
            const long long nid = idx ? (long long)idx[(long long)grow * fanout + t]
                                      : (long long)grow * fanout + t;
            if (srcIsBf16) {
                const v4bf v = *(const v4bf*)(srcB + nid * D_IN + c4);
                ax += (float)v[0]; ay += (float)v[1]; az += (float)v[2]; aw += (float)v[3];
            } else {
                const float4 v = *(const float4*)(srcF + nid * D_IN + c4);
                ax += v.x; ay += v.y; az += v.z; aw += v.w;
            }
        }
        v4bf o;
        o[0] = (__bf16)(ax * inv); o[1] = (__bf16)(ay * inv);
        o[2] = (__bf16)(az * inv); o[3] = (__bf16)(aw * inv);
        *(v4bf*)(&P[r * LDS_STRIDE + c4]) = o;
    }
    __syncthreads();

    // ---- WMMA: D[16x128] = A[16x256] x W^T, tiled 16x16x32 ----
    const int r  = lane & 15;   // row (A) / column (B) within tile
    const int hl = lane >> 4;   // lane half

#pragma unroll
    for (int nt = 0; nt < 8; ++nt) {
        v8f acc = {};
        const int n0 = nt * 16 + r;
#pragma unroll
        for (int kb = 0; kb < 8; ++kb) {
            // A fragment: lane r holds K = hl*16+0..15 of row r
            const v8bf alo = *(const v8bf*)(&P[r * LDS_STRIDE + kb * 32 + hl * 16]);
            const v8bf ahi = *(const v8bf*)(&P[r * LDS_STRIDE + kb * 32 + hl * 16 + 8]);
            const v16bf a = __builtin_shufflevector(alo, ahi,
                0, 1, 2, 3, 4, 5, 6, 7, 8, 9, 10, 11, 12, 13, 14, 15);
            // B fragment: lane n holds K = hl*16+0..15 of column n -> contiguous W row
            const v16bf b = *(const v16bf*)(W + (long)n0 * D_IN + kb * 32 + hl * 16);
            acc = __builtin_amdgcn_wmma_f32_16x16x32_bf16(
                false, a, false, b, (short)0, acc, false, false);
        }
        const float bv = bias[n0];
        if (rnorm == nullptr) {
            __bf16* o = (__bf16*)outv;
#pragma unroll
            for (int v = 0; v < 8; ++v) {
                const int grow = rowBase + v + 8 * hl;   // C/D layout: M = v + 8*(lane/16)
                const float val = fmaxf(acc[v] + bv, 0.0f);
                o[(long)grow * outStride + colBase + n0] = (__bf16)val;
            }
        } else {
            float* o = (float*)outv;
#pragma unroll
            for (int v = 0; v < 8; ++v) {
                const int grow = rowBase + v + 8 * hl;
                o[(long)grow * outStride + n0] = acc[v] * rnorm[grow] + bv;
            }
        }
    }
}

// ---------------------------------------------------------------------------
// rnorm[row] = 1 / max(||h_row||_2, 1e-12) ; one wave per row, wave32 butterfly
// ---------------------------------------------------------------------------
__global__ __launch_bounds__(256)
void rownorm_kernel(const __bf16* __restrict__ h, float* __restrict__ rnorm, int nrows) {
    const int row  = blockIdx.x * 8 + (threadIdx.x >> 5);
    const int lane = threadIdx.x & 31;
    if (row >= nrows) return;
    const v8bf v = *(const v8bf*)(h + (long)row * D_IN + lane * 8);
    float s = 0.f;
#pragma unroll
    for (int j = 0; j < 8; ++j) { const float f = (float)v[j]; s += f * f; }
#pragma unroll
    for (int off = 16; off > 0; off >>= 1) s += __shfl_xor(s, off, 32);
    if (lane == 0) rnorm[row] = 1.0f / fmaxf(sqrtf(s), 1e-12f);
}

// ---------------------------------------------------------------------------
extern "C" void kernel_launch(void* const* d_in, const int* in_sizes, int n_in,
                              void* d_out, int out_size, void* d_ws, size_t ws_size,
                              hipStream_t stream)
{
    const float* emb      = (const float*)d_in[0];
    const float* w_self0  = (const float*)d_in[1];
    const float* b_self0  = (const float*)d_in[2];
    const float* w_neigh0 = (const float*)d_in[3];
    const float* b_neigh0 = (const float*)d_in[4];
    const float* w_self1  = (const float*)d_in[5];
    const float* b_self1  = (const float*)d_in[6];
    const float* w_neigh1 = (const float*)d_in[7];
    const float* b_neigh1 = (const float*)d_in[8];
    const float* fc_w     = (const float*)d_in[9];
    const float* fc_b     = (const float*)d_in[10];
    const int*   nodeids  = (const int*)d_in[11];
    const int*   neigh1   = (const int*)d_in[12];
    const int*   neigh2   = (const int*)d_in[13];

    const int B  = in_sizes[11];   // 16384
    const int R1 = in_sizes[12];   // B * 3

    // ---- workspace carve-out (256B aligned) ----
    char*  ws  = (char*)d_ws;
    size_t off = 0;
    auto carve = [&](size_t bytes) {
        void* p = ws + off;
        off = (off + bytes + 255) & ~(size_t)255;
        return p;
    };
    const size_t WBYTES = (size_t)D_HALF * D_IN * sizeof(__bf16);  // 64 KB
    __bf16* wbf_s0 = (__bf16*)carve(WBYTES);
    __bf16* wbf_n0 = (__bf16*)carve(WBYTES);
    __bf16* wbf_s1 = (__bf16*)carve(WBYTES);
    __bf16* wbf_n1 = (__bf16*)carve(WBYTES);
    __bf16* wbf_fc = (__bf16*)carve(WBYTES);
    __bf16* g1     = (__bf16*)carve((size_t)R1 * D_IN * sizeof(__bf16)); // [B*3,256]
    __bf16* g0     = (__bf16*)carve((size_t)B  * D_IN * sizeof(__bf16)); // [B,256]
    __bf16* h      = (__bf16*)carve((size_t)B  * D_IN * sizeof(__bf16)); // [B,256]
    float*  rnorm  = (float*) carve((size_t)B * sizeof(float));
    (void)ws_size; (void)n_in; (void)out_size;

    // ---- weights f32 -> bf16 ----
    const int WN = D_HALF * D_IN;
    const dim3 cg((WN + 255) / 256), cb(256);
    cvt_bf16_kernel<<<cg, cb, 0, stream>>>(w_self0,  wbf_s0, WN);
    cvt_bf16_kernel<<<cg, cb, 0, stream>>>(w_neigh0, wbf_n0, WN);
    cvt_bf16_kernel<<<cg, cb, 0, stream>>>(w_self1,  wbf_s1, WN);
    cvt_bf16_kernel<<<cg, cb, 0, stream>>>(w_neigh1, wbf_n1, WN);
    cvt_bf16_kernel<<<cg, cb, 0, stream>>>(fc_w,     wbf_fc, WN);

    // ---- layer 0 on (feats1, feats2) -> g1 [B*3, 256] ----
    sage_half_wmma_kernel<<<R1 / 16, 32, 0, stream>>>(
        emb, 0, neigh1, 1, wbf_s0, b_self0, nullptr, g1, D_IN, 0);
    sage_half_wmma_kernel<<<R1 / 16, 32, 0, stream>>>(
        emb, 0, neigh2, 5, wbf_n0, b_neigh0, nullptr, g1, D_IN, D_HALF);

    // ---- layer 0 on (feats0, feats1) -> g0 [B, 256] ----
    sage_half_wmma_kernel<<<B / 16, 32, 0, stream>>>(
        emb, 0, nodeids, 1, wbf_s0, b_self0, nullptr, g0, D_IN, 0);
    sage_half_wmma_kernel<<<B / 16, 32, 0, stream>>>(
        emb, 0, neigh1, 3, wbf_n0, b_neigh0, nullptr, g0, D_IN, D_HALF);

    // ---- layer 1 on (g0, g1) -> h [B, 256] ----
    sage_half_wmma_kernel<<<B / 16, 32, 0, stream>>>(
        g0, 1, nullptr, 1, wbf_s1, b_self1, nullptr, h, D_IN, 0);
    sage_half_wmma_kernel<<<B / 16, 32, 0, stream>>>(
        g1, 1, nullptr, 3, wbf_n1, b_neigh1, nullptr, h, D_IN, D_HALF);

    // ---- row norms, then FC with folded normalization: (h W^T)/||h|| + b ----
    rownorm_kernel<<<B / 8, 256, 0, stream>>>(h, rnorm, B);
    sage_half_wmma_kernel<<<B / 16, 32, 0, stream>>>(
        h, 1, nullptr, 1, wbf_fc, fc_b, rnorm, d_out, D_HALF, 0);
}
